// MPNN_SenderAggr_Loop_40080634806382
// MI455X (gfx1250) — matrix operand order repaired and checked by reading, not compiled, not run on hardware
//
#include <hip/hip_runtime.h>

#define N_NODES_C 100000
#define N_EDGES_C 3200000
#define HDIM 32
#define WPB 4            // waves per block
#define BLOCK (WPB * 32)

typedef __attribute__((ext_vector_type(16))) _Float16 v16h;
typedef __attribute__((ext_vector_type(8)))  _Float16 h16x8;
typedef __attribute__((ext_vector_type(4)))  _Float16 h16x4;
typedef __attribute__((ext_vector_type(8)))  float    v8f;
typedef __attribute__((ext_vector_type(4)))  float    f32x4;
typedef __attribute__((ext_vector_type(2)))  float    f32x2;

// ---------------------------------------------------------------- WMMA helpers

static __device__ __forceinline__ v8f wmma16(v16h a, v16h b, v8f c) {
  return __builtin_amdgcn_wmma_f32_16x16x32_f16(false, a, false, b, (short)0, c,
                                                false, false);
}

static __device__ __forceinline__ void wave_sync() {
  __builtin_amdgcn_wave_barrier();  // scheduling fence; DS ops in-order per wave
}

static __device__ __forceinline__ v8f splat8(float v) {
  v8f c;
#pragma unroll
  for (int i = 0; i < 8; ++i) c[i] = v;
  return c;
}

static __device__ __forceinline__ v8f lrelu8(v8f x) {
#pragma unroll
  for (int i = 0; i < 8; ++i) x[i] = fmaxf(x[i], 0.01f * x[i]);
  return x;
}

// A fragment (16x32 f16) gathered from row-major f32 [*,32]; row = lane&15's
// gather target. Two 32B chunks per lane -> 4x global_load_b128.
template <bool NT>
static __device__ __forceinline__ v16h a_frag_g(const float* __restrict__ base,
                                                unsigned row, int lane) {
  const f32x4* r4 = (const f32x4*)(base + row * HDIM) + (lane >> 4) * 2;
  f32x4 q0, q1, q2, q3;
  if (NT) {
    q0 = __builtin_nontemporal_load(r4 + 0);
    q1 = __builtin_nontemporal_load(r4 + 1);
    q2 = __builtin_nontemporal_load(r4 + 4);
    q3 = __builtin_nontemporal_load(r4 + 5);
  } else {
    q0 = r4[0]; q1 = r4[1]; q2 = r4[4]; q3 = r4[5];
  }
  v16h a;
#pragma unroll
  for (int i = 0; i < 4; ++i) {
    a[i]      = (_Float16)q0[i];
    a[i + 4]  = (_Float16)q1[i];
    a[i + 8]  = (_Float16)q2[i];
    a[i + 12] = (_Float16)q3[i];
  }
  return a;
}

// A fragment from LDS f16 tile (stride in halves, rows 16B aligned): 2x ds_load_b128
static __device__ __forceinline__ v16h a_frag_lds(const _Float16* t, int stride,
                                                  int kbase, int lane) {
  const _Float16* pb = t + (lane & 15) * stride + kbase + (lane >> 4) * 8;
  const h16x8 lo = *(const h16x8*)pb;
  const h16x8 hi = *(const h16x8*)(pb + 16);
  v16h a;
#pragma unroll
  for (int i = 0; i < 8; ++i) { a[i] = lo[i]; a[i + 8] = hi[i]; }
  return a;
}

// B fragment (32x16 f16) from row-major f16 weight [K][32] in LDS (hoisted).
static __device__ __forceinline__ v16h b_frag(const _Float16* w, int kbase,
                                              int nhalf, int lane) {
  const int col = (lane & 15) + nhalf * 16;
  const int kh  = (lane >> 4) * 8 + kbase;
  v16h b;
#pragma unroll
  for (int v = 0; v < 8; ++v) {
    const int k = kh + 2 * v + (v >= 4 ? 8 : 0);
    b[2 * v]     = w[k * HDIM + col];
    b[2 * v + 1] = w[(k + 1) * HDIM + col];
  }
  return b;
}

// D layout: lane holds col N=(lane&15)[+16*nhalf], rows M=(lane>>4)*8 + r.
static __device__ __forceinline__ void d_to_lds_f16(_Float16* t, int stride,
                                                    int cbase, v8f d, int nhalf,
                                                    int lane) {
  const int col   = cbase + (lane & 15) + nhalf * 16;
  const int mbase = (lane >> 4) * 8;
#pragma unroll
  for (int r = 0; r < 8; ++r) t[(mbase + r) * stride + col] = (_Float16)d[r];
}

static __device__ __forceinline__ void d_to_lds_f32(float* t, v8f d, int nhalf,
                                                    int lane) {
  const int col   = (lane & 15) + nhalf * 16;
  const int mbase = (lane >> 4) * 8;
#pragma unroll
  for (int r = 0; r < 8; ++r) t[(mbase + r) * HDIM + col] = d[r];
}

// Scatter-add D frag into acc[idx*32+col]; 32-bit offsets -> SADDR atomics.
static __device__ __forceinline__ void scatter_add(float* __restrict__ acc,
                                                   v8f d, int nhalf, int lane,
                                                   int myIdx) {
  const int col   = (lane & 15) + nhalf * 16;
  const int mbase = (lane >> 4) * 8;
#pragma unroll
  for (int r = 0; r < 8; ++r) {
    const unsigned di = (unsigned)__shfl(myIdx, mbase + r, 32);
    atomicAdd(acc + (di * HDIM + (unsigned)col), d[r]);
  }
}

// ---------------------------------------------------------------- small kernels

__global__ void k_zero4(f32x4* __restrict__ p, int n4) {
  const int i = blockIdx.x * blockDim.x + threadIdx.x;
  if (i < n4) p[i] = (f32x4)0.f;
}

// h_node = x @ W_in + b_in   (x: [N,3])
__global__ void k_hnode(const float* __restrict__ x,
                        const float* __restrict__ W_in,
                        const float* __restrict__ b_in,
                        float* __restrict__ h_node) {
  const unsigned i = blockIdx.x * blockDim.x + threadIdx.x;
  if (i >= (unsigned)N_NODES_C * HDIM) return;
  const unsigned n = i >> 5;
  const int c = (int)(i & 31);
  const float* xr = x + n * 3;
  h_node[i] = b_in[c] + xr[0] * W_in[c] + xr[1] * W_in[HDIM + c] +
              xr[2] * W_in[2 * HDIM + c];
}

// ---------------------------------------------------------------- phase 1
// aggr_msgs[dst] += (h_msg @ We + be).  encoded is recomputed in phase 2
// (saves a 410MB spill); accumulators are L2-resident, h_msg is streamed NT.
__global__ void k_phase1(const float* __restrict__ h_msg,
                         const long long* __restrict__ eidx,
                         const float* __restrict__ We,
                         const float* __restrict__ be,
                         float* __restrict__ aggr_msgs) {
  __shared__ __align__(16) _Float16 sWe[HDIM * HDIM];
  for (int i = threadIdx.x; i < HDIM * HDIM; i += blockDim.x)
    sWe[i] = (_Float16)We[i];
  __syncthreads();

  const int lane = threadIdx.x & 31;
  const int wave = threadIdx.x >> 5;
  const long long* dstI = eidx + N_EDGES_C;
  const int nTiles = N_EDGES_C / 16;

  const v16h b0 = b_frag(sWe, 0, 0, lane);
  const v16h b1 = b_frag(sWe, 0, 1, lane);
  const float be0 = be[lane & 15], be1 = be[(lane & 15) + 16];

  for (int tile = blockIdx.x * WPB + wave; tile < nTiles;
       tile += gridDim.x * WPB) {
    const unsigned e = (unsigned)tile * 16 + (lane & 15);
    const v16h a = a_frag_g<true>(h_msg, e, lane);
    const v8f c0 = wmma16(a, b0, splat8(be0));
    const v8f c1 = wmma16(a, b1, splat8(be1));
    const int myDst = (int)dstI[e];
    scatter_add(aggr_msgs, c0, 0, lane, myDst);
    scatter_add(aggr_msgs, c1, 1, lane, myDst);
  }
}

// ---------------------------------------------------------------- phase 2
__global__ void k_phase2(const float* __restrict__ h_msg,
                         const long long* __restrict__ eidx,
                         const float* __restrict__ h_node,
                         const float* __restrict__ aggr_msgs,
                         const float* We,  const float* be,
                         const float* Wn1, const float* bn1,
                         const float* Wn2, const float* bn2,
                         const float* WN1, const float* bN1,
                         const float* WN2, const float* bN2,
                         const float* Wd,  const float* bd,
                         float* __restrict__ out_hmsg,
                         float* __restrict__ out_ymsg,
                         float* __restrict__ aggr_out) {
  __shared__ __align__(16) _Float16 sWe[1024], sWn1[1024], sWn2[1024],
                                    sWN2[1024], sWN1[2048];
  __shared__ __align__(16) float    sWd[64];
  __shared__ __align__(16) _Float16 sInter[WPB][16 * 40];  // 80B rows
  __shared__ __align__(16) _Float16 sCat[WPB][16 * 72];    // 144B rows
  __shared__ __align__(16) float    sOut[WPB][16 * 32];

  for (int i = threadIdx.x; i < 1024; i += blockDim.x) {
    sWe[i]  = (_Float16)We[i];
    sWn1[i] = (_Float16)Wn1[i];
    sWn2[i] = (_Float16)Wn2[i];
    sWN2[i] = (_Float16)WN2[i];
  }
  for (int i = threadIdx.x; i < 2048; i += blockDim.x)
    sWN1[i] = (_Float16)WN1[i];
  for (int i = threadIdx.x; i < 64; i += blockDim.x) sWd[i] = Wd[i];
  __syncthreads();

  const int lane  = threadIdx.x & 31;
  const int wave  = threadIdx.x >> 5;
  const int row   = lane & 15;
  const int khalf = lane >> 4;
  const long long* srcI = eidx;
  const long long* dstI = eidx + N_EDGES_C;
  _Float16* inter = sInter[wave];
  _Float16* cat   = sCat[wave];
  float*    otile = sOut[wave];
  const int nTiles = N_EDGES_C / 16;

  // Hoist all B fragments and per-lane bias scalars out of the tile loop.
  const v16h bWe0  = b_frag(sWe,  0,  0, lane), bWe1  = b_frag(sWe,  0,  1, lane);
  const v16h bn1f0 = b_frag(sWn1, 0,  0, lane), bn1f1 = b_frag(sWn1, 0,  1, lane);
  const v16h bn2f0 = b_frag(sWn2, 0,  0, lane), bn2f1 = b_frag(sWn2, 0,  1, lane);
  const v16h bN1f00 = b_frag(sWN1, 0,  0, lane), bN1f10 = b_frag(sWN1, 32, 0, lane);
  const v16h bN1f01 = b_frag(sWN1, 0,  1, lane), bN1f11 = b_frag(sWN1, 32, 1, lane);
  const v16h bN2f0 = b_frag(sWN2, 0,  0, lane), bN2f1 = b_frag(sWN2, 0,  1, lane);
  const int c0i = lane & 15, c1i = (lane & 15) + 16;
  const float vbe0 = be[c0i],  vbe1 = be[c1i];
  const float vb10 = bn1[c0i], vb11 = bn1[c1i];
  const float vb20 = bn2[c0i], vb21 = bn2[c1i];
  const float vN10 = bN1[c0i], vN11 = bN1[c1i];
  const float vN20 = bN2[c0i], vN21 = bN2[c1i];
  const float bd0 = bd[0], bd1 = bd[1];

  for (int tile = blockIdx.x * WPB + wave; tile < nTiles;
       tile += gridDim.x * WPB) {
    const unsigned base16 = (unsigned)tile * 16;
    const unsigned e      = base16 + row;
    const int mySrc = (int)srcI[e];
    const int myDst = (int)dstI[e];

    // cat[:, 0:32) = aggr_msgs[src]  (L2 gather, packed f16 b64 stores)
    {
      const f32x4* p4 = (const f32x4*)(aggr_msgs + (unsigned)mySrc * HDIM) +
                        khalf * 4;
      h16x4* c4 = (h16x4*)(cat + row * 72 + khalf * 16);
#pragma unroll
      for (int j = 0; j < 4; ++j) {
        const f32x4 q = p4[j];
        h16x4 h;
#pragma unroll
        for (int i = 0; i < 4; ++i) h[i] = (_Float16)q[i];
        c4[j] = h;
      }
    }

    // encoded = h_msg @ We + be ; cat[:, 32:64) = encoded
    const v16h am = a_frag_g<true>(h_msg, e, lane);
    const v8f enc0 = wmma16(am, bWe0, splat8(vbe0));
    const v8f enc1 = wmma16(am, bWe1, splat8(vbe1));
    d_to_lds_f16(cat, 72, 32, enc0, 0, lane);
    d_to_lds_f16(cat, 72, 32, enc1, 1, lane);

    // nn_out = lrelu(lrelu(h_node[src] @ Wn1 + bn1) @ Wn2 + bn2)
    const v16h ah = a_frag_g<false>(h_node, (unsigned)mySrc, lane);
    const v8f t0 = lrelu8(wmma16(ah, bn1f0, splat8(vb10)));
    const v8f t1 = lrelu8(wmma16(ah, bn1f1, splat8(vb11)));
    d_to_lds_f16(inter, 40, 0, t0, 0, lane);
    d_to_lds_f16(inter, 40, 0, t1, 1, lane);
    wave_sync();
    const v16h ai = a_frag_lds(inter, 40, 0, lane);
    const v8f nn0 = lrelu8(wmma16(ai, bn2f0, splat8(vb20)));
    const v8f nn1 = lrelu8(wmma16(ai, bn2f1, splat8(vb21)));

    // n_out = lrelu(lrelu([aggr_msgs[src], encoded] @ WN1 + bN1) @ WN2 + bN2)
    wave_sync();
    const v16h ac0 = a_frag_lds(cat, 72, 0, lane);
    const v16h ac1 = a_frag_lds(cat, 72, 32, lane);
    v8f u0 = splat8(vN10);
    u0 = wmma16(ac0, bN1f00, u0);
    u0 = wmma16(ac1, bN1f10, u0);
    u0 = lrelu8(u0);
    v8f u1 = splat8(vN11);
    u1 = wmma16(ac0, bN1f01, u1);
    u1 = wmma16(ac1, bN1f11, u1);
    u1 = lrelu8(u1);
    d_to_lds_f16(inter, 40, 0, u0, 0, lane);
    d_to_lds_f16(inter, 40, 0, u1, 1, lane);
    wave_sync();
    const v16h au = a_frag_lds(inter, 40, 0, lane);
    const v8f n0 = lrelu8(wmma16(au, bN2f0, splat8(vN20)));
    const v8f n1 = lrelu8(wmma16(au, bN2f1, splat8(vN21)));

    // h_msg_new = nn_out + n_out
    v8f m0, m1;
#pragma unroll
    for (int i = 0; i < 8; ++i) { m0[i] = nn0[i] + n0[i]; m1[i] = nn1[i] + n1[i]; }

    // scatter into aggr_out[dst]  (L2-resident atomics, 32-bit offsets)
    scatter_add(aggr_out, m0, 0, lane, myDst);
    scatter_add(aggr_out, m1, 1, lane, myDst);

    // stage f32 tile; NT-coalesced b128 stores of h_msg_new
    d_to_lds_f32(otile, m0, 0, lane);
    d_to_lds_f32(otile, m1, 1, lane);
    wave_sync();
    {
      f32x4* op = (f32x4*)out_hmsg + (unsigned)tile * 128u;
#pragma unroll
      for (int j = 0; j < 4; ++j) {
        const int t = j * 32 + lane;
        __builtin_nontemporal_store(((const f32x4*)otile)[t], op + t);
      }
    }

    // y_msg = softmax(h_msg_new @ Wd + bd): both lane-halves + shfl_xor reduce
    {
      const float* hr = otile + row * HDIM + khalf * 16;
      float l0 = 0.f, l1 = 0.f;
#pragma unroll
      for (int k = 0; k < 16; ++k) {
        const float h = hr[k];
        l0 += h * sWd[(khalf * 16 + k) * 2 + 0];
        l1 += h * sWd[(khalf * 16 + k) * 2 + 1];
      }
      l0 += __shfl_xor(l0, 16, 32);
      l1 += __shfl_xor(l1, 16, 32);
      if (lane < 16) {
        l0 += bd0; l1 += bd1;
        const float mx = fmaxf(l0, l1);
        const float e0 = __expf(l0 - mx), e1 = __expf(l1 - mx);
        const float inv = 1.0f / (e0 + e1);
        f32x2 y; y[0] = e0 * inv; y[1] = e1 * inv;
        __builtin_nontemporal_store(y, (f32x2*)out_ymsg + (base16 + lane));
      }
    }
    wave_sync();
  }
}

// ---------------------------------------------------------------- phase 3
// beliefs (first N/2 nodes): softmax(lrelu([h_node,aggr_out]@WU+bU) @ Wb + bb)
__global__ void k_phase3(const float* __restrict__ h_node,
                         const float* __restrict__ aggr_out,
                         const float* WU, const float* bU,
                         const float* Wb, const float* bb,
                         float* __restrict__ out_bel) {
  __shared__ __align__(16) _Float16 sWU[2048];
  __shared__ __align__(16) float sWb[64];
  __shared__ __align__(16) float sOut[WPB][16 * 32];
  for (int i = threadIdx.x; i < 2048; i += blockDim.x)
    sWU[i] = (_Float16)WU[i];
  for (int i = threadIdx.x; i < 64; i += blockDim.x) sWb[i] = Wb[i];
  __syncthreads();

  const int lane  = threadIdx.x & 31;
  const int wave  = threadIdx.x >> 5;
  const int row   = lane & 15;
  const int khalf = lane >> 4;
  float* otile = sOut[wave];
  const int nTiles = (N_NODES_C / 2) / 16;  // 3125

  const v16h bU00 = b_frag(sWU, 0, 0, lane), bU10 = b_frag(sWU, 32, 0, lane);
  const v16h bU01 = b_frag(sWU, 0, 1, lane), bU11 = b_frag(sWU, 32, 1, lane);
  const float vU0 = bU[lane & 15], vU1 = bU[(lane & 15) + 16];
  const float bb0 = bb[0], bb1 = bb[1];

  for (int tile = blockIdx.x * WPB + wave; tile < nTiles;
       tile += gridDim.x * WPB) {
    const unsigned r = (unsigned)tile * 16 + row;
    const v16h a0 = a_frag_g<false>(h_node, r, lane);
    const v16h a1 = a_frag_g<false>(aggr_out, r, lane);
    v8f c0 = splat8(vU0);
    c0 = wmma16(a0, bU00, c0);
    c0 = wmma16(a1, bU10, c0);
    c0 = lrelu8(c0);
    v8f c1 = splat8(vU1);
    c1 = wmma16(a0, bU01, c1);
    c1 = wmma16(a1, bU11, c1);
    c1 = lrelu8(c1);
    d_to_lds_f32(otile, c0, 0, lane);
    d_to_lds_f32(otile, c1, 1, lane);
    wave_sync();
    {
      const float* hr = otile + row * HDIM + khalf * 16;
      float l0 = 0.f, l1 = 0.f;
#pragma unroll
      for (int k = 0; k < 16; ++k) {
        const float h = hr[k];
        l0 += h * sWb[(khalf * 16 + k) * 2 + 0];
        l1 += h * sWb[(khalf * 16 + k) * 2 + 1];
      }
      l0 += __shfl_xor(l0, 16, 32);
      l1 += __shfl_xor(l1, 16, 32);
      if (lane < 16) {
        l0 += bb0; l1 += bb1;
        const float mx = fmaxf(l0, l1);
        const float e0 = __expf(l0 - mx), e1 = __expf(l1 - mx);
        const float inv = 1.0f / (e0 + e1);
        f32x2 y; y[0] = e0 * inv; y[1] = e1 * inv;
        __builtin_nontemporal_store(y,
            (f32x2*)out_bel + ((unsigned)tile * 16 + lane));
      }
    }
    wave_sync();
  }
}

// ---------------------------------------------------------------- launch

extern "C" void kernel_launch(void* const* d_in, const int* in_sizes, int n_in,
                              void* d_out, int out_size, void* d_ws,
                              size_t ws_size, hipStream_t stream) {
  (void)in_sizes; (void)n_in; (void)out_size; (void)ws_size;
  const float*     x    = (const float*)d_in[0];
  const long long* ei   = (const long long*)d_in[1];  // int64 [2,E]
  const float* h_msg = (const float*)d_in[2];
  const float* W_in  = (const float*)d_in[3];
  const float* b_in  = (const float*)d_in[4];
  const float* We  = (const float*)d_in[5];
  const float* be  = (const float*)d_in[6];
  const float* Wn1 = (const float*)d_in[7];
  const float* bn1 = (const float*)d_in[8];
  const float* Wn2 = (const float*)d_in[9];
  const float* bn2 = (const float*)d_in[10];
  const float* WN1 = (const float*)d_in[11];
  const float* bN1 = (const float*)d_in[12];
  const float* WN2 = (const float*)d_in[13];
  const float* bN2 = (const float*)d_in[14];
  const float* WU  = (const float*)d_in[15];
  const float* bU  = (const float*)d_in[16];
  const float* Wd  = (const float*)d_in[17];
  const float* bd  = (const float*)d_in[18];
  const float* Wb  = (const float*)d_in[19];
  const float* bb  = (const float*)d_in[20];

  float* ws        = (float*)d_ws;
  float* h_node    = ws;                                 // [N,32]
  float* aggr_msgs = ws + (size_t)N_NODES_C * HDIM;      // [N,32]
  float* aggr_out  = ws + (size_t)N_NODES_C * HDIM * 2;  // [N,32]

  float* out_hmsg = (float*)d_out;                        // [E,32]
  float* out_ymsg = out_hmsg + (size_t)N_EDGES_C * HDIM;  // [E,2]
  float* out_bel  = out_ymsg + (size_t)N_EDGES_C * 2;     // [N/2,2]

  const int nz4 = N_NODES_C * (HDIM / 4) * 2;  // both accumulators, float4s
  k_zero4<<<(nz4 + 255) / 256, 256, 0, stream>>>((f32x4*)aggr_msgs, nz4);

  const int nh = N_NODES_C * HDIM;
  k_hnode<<<(nh + 255) / 256, 256, 0, stream>>>(x, W_in, b_in, h_node);

  k_phase1<<<4096, BLOCK, 0, stream>>>(h_msg, ei, We, be, aggr_msgs);

  k_phase2<<<8192, BLOCK, 0, stream>>>(h_msg, ei, h_node, aggr_msgs,
                                       We, be, Wn1, bn1, Wn2, bn2,
                                       WN1, bN1, WN2, bN2, Wd, bd,
                                       out_hmsg, out_ymsg, aggr_out);

  const int nTiles3 = (N_NODES_C / 2) / 16;
  k_phase3<<<(nTiles3 + WPB - 1) / WPB, BLOCK, 0, stream>>>(
      h_node, aggr_out, WU, bU, Wb, bb, out_bel);
}
